// NADE_10118942949654
// MI455X (gfx1250) — matrix operand order repaired and checked by reading, not compiled, not run on hardware
//
#include <hip/hip_runtime.h>
#include <math.h>
#include <stdint.h>

typedef __attribute__((ext_vector_type(16))) _Float16 v16h;
typedef __attribute__((ext_vector_type(8)))  float    v8f;

#define B_BATCH 512
#define D_DIM   784
#define H_DIM   500
#define CHUNK   32
#define NCHUNK  25            // 25*32 = 800 >= 784 (tail zero-padded)
#define XPAD_D  800
#define ROWS_PB 16
#define THREADS 256           // 8 waves; each wave owns 2 batch rows
#define NSLICE  4             // independent h-slices (relu is per-h; l is additive over h)
#define HT_SL   8             // h-tiles per slice
#define H_SL    128           // h per slice

// fragment-array element counts
#define WCF_CHUNK 16384       // halves per chunk: 32 ht * 32 lane * 16 slots
#define VCF_CHUNK 16384       // floats per chunk: 32 ht * 2 rg * 32 lane * 8 slots
#define WCF_SL    4096        // per chunk-slice (8 ht)
#define VCF_SL    4096

// ws layout (bytes): VcF f32 | WcF f16 | xPad f32 | Lpartial f32  (~10.5 MB)
#define WS_VCF_OFF 0
#define WS_WCF_OFF (NCHUNK * VCF_CHUNK * 4)                       // 1638400
#define WS_XP_OFF  (WS_WCF_OFF + NCHUNK * WCF_CHUNK * 2)          // 2457600
#define WS_LP_OFF  (WS_XP_OFF + B_BATCH * XPAD_D * 4)             // 4096000
#define WS_TOTAL   (WS_LP_OFF + NSLICE * B_BATCH * D_DIM * 4)     // 10518528

// LDS layout (bytes): aBase | VcL[2] | xcL[2] | WcL[2]  = 60KB/block
#define AB_B  0
#define VC_B  (AB_B + ROWS_PB * H_SL * 4)         // 8192
#define XC_B  (VC_B + 2 * VCF_SL * 4)             // 40960
#define WC_B  (XC_B + 2 * ROWS_PB * CHUNK * 4)    // 45056
#define LDS_TOTAL (WC_B + 2 * WCF_SL * 2)         // 61440

// ---------------- prep: swizzle W/V/x into WMMA-fragment order ----------------
__global__ __launch_bounds__(THREADS)
void nade_prep_kernel(const float* __restrict__ W, const float* __restrict__ V,
                      const float* __restrict__ x, char* __restrict__ ws)
{
    _Float16* WcF = (_Float16*)(ws + WS_WCF_OFF);
    float*    VcF = (float*)(ws + WS_VCF_OFF);
    float*    xP  = (float*)(ws + WS_XP_OFF);
    int idx = blockIdx.x * THREADS + threadIdx.x;   // 0 .. 409599

    // WcF[ck][ht][lane][i] = f16(W[h][d]),  k=(lane>>4)*16+i, h=ht*16+(lane&15)
    {
        int i = idx & 15, l = (idx >> 4) & 31, ht = (idx >> 9) & 31, ck = idx >> 14;
        int k = (l >> 4) * 16 + i;
        int h = ht * 16 + (l & 15);
        int d = ck * CHUNK + k;
        float v = (h < H_DIM && d < D_DIM) ? W[h * D_DIM + d] : 0.f;
        WcF[idx] = (_Float16)v;
    }
    // VcF[ck][ht][rg][lane][rr] = V[d][h],  d = ck*32+rg*16+8*(lane>>4)+rr
    {
        int rr = idx & 7, l = (idx >> 3) & 31, rg = (idx >> 8) & 1,
            ht = (idx >> 9) & 31, ck = idx >> 14;
        int d = ck * CHUNK + rg * 16 + 8 * (l >> 4) + rr;
        int h = ht * 16 + (l & 15);
        VcF[idx] = (d < D_DIM && h < H_DIM) ? V[d * H_DIM + h] : 0.f;
    }
    // xPad[b][0..799] = x zero-padded to 800 cols
    {
        int b = idx / XPAD_D, d = idx % XPAD_D;
        xP[b * XPAD_D + d] = (d < D_DIM) ? x[b * D_DIM + d] : 0.f;
    }
}

// ---------------- async helpers (CDNA5 ASYNCcnt path) ----------------
__device__ __forceinline__ void async_b128(uint32_t lds_byte, uint64_t sbase, uint32_t voff)
{
    asm volatile("global_load_async_to_lds_b128 %0, %1, %2"
                 :: "v"(lds_byte), "v"(voff), "s"(sbase) : "memory");
}
__device__ __forceinline__ void wait_async0()
{
    asm volatile("s_wait_asynccnt 0x0" ::: "memory");
}

// single-instruction 16-lane xor-shuffle reduction (ds_swizzle group-of-32)
__device__ __forceinline__ float red16_xor(float s)
{
    s += __int_as_float(__builtin_amdgcn_ds_swizzle(__float_as_int(s), 0x041f)); // xor 1
    s += __int_as_float(__builtin_amdgcn_ds_swizzle(__float_as_int(s), 0x081f)); // xor 2
    s += __int_as_float(__builtin_amdgcn_ds_swizzle(__float_as_int(s), 0x101f)); // xor 4
    s += __int_as_float(__builtin_amdgcn_ds_swizzle(__float_as_int(s), 0x201f)); // xor 8
    return s;
}

__device__ __forceinline__ void issue_chunk(int buf, int ck, int sl, int r0, int tid,
                                            uint64_t vb, uint64_t wb, uint64_t xb)
{
    // VcF chunk-slice: 16KB
    uint32_t vl = VC_B + buf * (VCF_SL * 4);
    uint32_t vs = (uint32_t)((ck * VCF_CHUNK + sl * VCF_SL) * 4);
    #pragma unroll
    for (int o = 0; o < VCF_SL; o += THREADS * 4) {
        uint32_t e = (o + tid * 4) * 4;
        async_b128(vl + e, vb, vs + e);
    }
    // WcF chunk-slice: 8KB
    uint32_t wl = WC_B + buf * (WCF_SL * 2);
    uint32_t wsrc = (uint32_t)((ck * WCF_CHUNK + sl * WCF_SL) * 2);
    #pragma unroll
    for (int o = 0; o < WCF_SL; o += THREADS * 8) {
        uint32_t e = (o + tid * 8) * 2;
        async_b128(wl + e, wb, wsrc + e);
    }
    // x tile: 16 rows x 32 floats (contiguous per row in xPad)
    if (tid < 128) {
        int bi = tid >> 3, seg = tid & 7;
        uint32_t xl = XC_B + buf * (ROWS_PB * CHUNK * 4) + (bi * CHUNK + seg * 4) * 4;
        uint32_t xs = (uint32_t)(((r0 + bi) * XPAD_D + ck * CHUNK + seg * 4) * 4);
        async_b128(xl, xb, xs);
    }
}

// ---------------- main kernel: 32 row-tiles x 4 h-slices ----------------
__global__ __launch_bounds__(THREADS)
void nade_wmma_main(const char* __restrict__ ws, const float* __restrict__ c,
                    float* __restrict__ Lp)
{
    extern __shared__ char smem[];
    float*    aBase = (float*)(smem + AB_B);     // [16][128] slice accumulator
    float*    VcL   = (float*)(smem + VC_B);     // [2][VCF_SL]
    float*    xcL   = (float*)(smem + XC_B);     // [2][16*32]
    _Float16* WcL   = (_Float16*)(smem + WC_B);  // [2][WCF_SL]

    const int tid  = threadIdx.x;
    const int lane = tid & 31;
    const int wave = tid >> 5;
    const int g    = lane >> 4;
    const int ln   = lane & 15;
    const int rb   = blockIdx.x;          // row-tile
    const int sl   = blockIdx.y;          // h-slice
    const int r0   = rb * ROWS_PB;
    const int bi0  = wave * 2;
    const int bg0  = r0 + bi0;

    const uint64_t vb = (uint64_t)(uintptr_t)(ws + WS_VCF_OFF);
    const uint64_t wb = (uint64_t)(uintptr_t)(ws + WS_WCF_OFF);
    const uint64_t xb = (uint64_t)(uintptr_t)(ws + WS_XP_OFF);

    // prefetch chunk 0 while initializing aBase = c-slice
    issue_chunk(0, 0, sl, r0, tid, vb, wb, xb);
    for (int idx = tid; idx < ROWS_PB * H_SL; idx += THREADS) {
        int h = sl * H_SL + (idx & (H_SL - 1));
        aBase[idx] = (h < H_DIM) ? c[h] : 0.f;
    }
    wait_async0();
    __syncthreads();

    int cur = 0;
    #pragma clang loop unroll(disable)
    for (int ck = 0; ck < NCHUNK; ++ck) {
        const int d0 = ck * CHUNK;
        if (ck + 1 < NCHUNK)
            issue_chunk(cur ^ 1, ck + 1, sl, r0, tid, vb, wb, xb);

        const float*    Vcur = VcL + cur * VCF_SL;
        const _Float16* Wcur = WcL + cur * WCF_SL;
        const float*    xcur = xcL + cur * (ROWS_PB * CHUNK);

        // triangular A-fragments: A[m][k] = (k < m) ? x[k] : 0  (built once/chunk)
        v16h  af[2][2];
        float x31[2];
        #pragma unroll
        for (int b = 0; b < 2; ++b) {
            const float* xr = &xcur[(bi0 + b) * CHUNK];
            x31[b] = xr[31];
            #pragma unroll
            for (int rg = 0; rg < 2; ++rg) {
                int m = rg * 16 + ln;
                #pragma unroll
                for (int i = 0; i < 16; ++i) {
                    int k = g * 16 + i;
                    af[b][rg][i] = (k < m) ? (_Float16)xr[k] : (_Float16)0.f;
                }
            }
        }

        float acc[2][2][8];
        #pragma unroll
        for (int b = 0; b < 2; ++b)
            #pragma unroll
            for (int rg = 0; rg < 2; ++rg)
                #pragma unroll
                for (int rr = 0; rr < 8; ++rr) acc[b][rg][rr] = 0.f;

        // keep as a real loop: full unroll blows past 256 VGPRs (s_set_vgpr_msb)
        #pragma clang loop unroll(disable)
        for (int htl = 0; htl < HT_SL; ++htl) {
            const int hb = htl * 16 + ln;   // slice-local h
            v16h bf  = *(const v16h*)(Wcur + (htl * 32 + lane) * 16);
            v8f  vv0 = *(const v8f*)(Vcur + ((htl * 2 + 0) * 32 + lane) * 8);
            v8f  vv1 = *(const v8f*)(Vcur + ((htl * 2 + 1) * 32 + lane) * 8);
            float w31 = (float)bf[15];      // K=31 sits in slot 15 of g==1 lanes

            #pragma unroll
            for (int b = 0; b < 2; ++b) {
                float aOld = aBase[(bi0 + b) * H_SL + hb];
                #pragma unroll
                for (int rg = 0; rg < 2; ++rg) {
                    v8f cz = {};
                    v8f dv = __builtin_amdgcn_wmma_f32_16x16x32_f16(
                        false, af[b][rg], false, bf, (short)0, cz, false, false);
                    #pragma unroll
                    for (int rr = 0; rr < 8; ++rr) {
                        float aval = aOld + dv[rr];
                        float hr   = fmaxf(aval, 0.f);
                        acc[b][rg][rr] += hr * (rg ? vv1[rr] : vv0[rr]);
                    }
                    if (rg == 1 && g == 1)  // fold full-chunk rank-32 update
                        aBase[(bi0 + b) * H_SL + hb] = aOld + dv[7] + x31[b] * w31;
                }
            }
        }

        // 16-lane reductions -> per-slice partial L
        #pragma unroll
        for (int b = 0; b < 2; ++b) {
            #pragma unroll
            for (int rg = 0; rg < 2; ++rg) {
                #pragma unroll
                for (int rr = 0; rr < 8; ++rr) {
                    float s = red16_xor(acc[b][rg][rr]);
                    if (ln == 0) {
                        int d = d0 + rg * 16 + rr + 8 * g;
                        if (d < D_DIM)
                            Lp[((size_t)sl * B_BATCH + bg0 + b) * D_DIM + d] = s;
                    }
                }
            }
        }

        wait_async0();
        __syncthreads();
        cur ^= 1;
    }
}

// ---------------- finalize: sum slices, bias, sigmoid, sample ----------------
__global__ __launch_bounds__(THREADS)
void nade_finalize(const char* __restrict__ ws, const float* __restrict__ u,
                   const float* __restrict__ bvec,
                   float* __restrict__ out_l, float* __restrict__ out_xs)
{
    int idx = blockIdx.x * THREADS + threadIdx.x;
    if (idx >= B_BATCH * D_DIM) return;
    int d = idx % D_DIM;
    const float* Lp = (const float*)(ws + WS_LP_OFF);
    float l = bvec[d];
    #pragma unroll
    for (int s = 0; s < NSLICE; ++s)
        l += Lp[(size_t)s * B_BATCH * D_DIM + idx];
    out_l[idx] = l;
    float p = 1.f / (1.f + __expf(-l));
    out_xs[idx] = (u[idx] < p) ? 1.f : 0.f;
}

extern "C" void kernel_launch(void* const* d_in, const int* in_sizes, int n_in,
                              void* d_out, int out_size, void* d_ws, size_t ws_size,
                              hipStream_t stream) {
    (void)in_sizes; (void)n_in; (void)out_size; (void)ws_size; // assumes ws_size >= ~10.5 MB
    const float* x = (const float*)d_in[0];
    const float* u = (const float*)d_in[1];
    const float* W = (const float*)d_in[2];
    const float* c = (const float*)d_in[3];
    const float* V = (const float*)d_in[4];
    const float* b = (const float*)d_in[5];
    float* out_l  = (float*)d_out;                      // l first, then x_sample
    float* out_xs = out_l + (size_t)B_BATCH * D_DIM;
    char*  ws     = (char*)d_ws;
    float* Lp     = (float*)(ws + WS_LP_OFF);

    // 1) one-time swizzle of W/V/x into WMMA-fragment order
    nade_prep_kernel<<<dim3((NCHUNK * VCF_CHUNK) / THREADS), dim3(THREADS), 0, stream>>>(
        W, V, x, ws);

    // 2) fused triangular-WMMA scan: 32 row-tiles x 4 h-slices, async double-buffered
    nade_wmma_main<<<dim3(B_BATCH / ROWS_PB, NSLICE), dim3(THREADS), LDS_TOTAL, stream>>>(
        ws, c, Lp);

    // 3) combine slice partials, bias, sigmoid, Bernoulli sample
    nade_finalize<<<dim3((B_BATCH * D_DIM + THREADS - 1) / THREADS), dim3(THREADS), 0, stream>>>(
        ws, u, b, out_l, out_xs);
}
